// GCN_24223615549761
// MI455X (gfx1250) — matrix operand order repaired and checked by reading, not compiled, not run on hardware
//
#include <hip/hip_runtime.h>
#include <hip/hip_bf16.h>

// GCN: out = spmm(relu(spmm(x) @ W0)) @ W1, with D^-1/2 (A + A^T + I) D^-1/2

#define NN 50000
#define NE 800000
#define DF 128      // D_FEAT == D_HID
#define NC 64       // N_CLASSES

typedef __attribute__((ext_vector_type(16))) __bf16 v16bf;
typedef __attribute__((ext_vector_type(8)))  __bf16 v8bf;
typedef __attribute__((ext_vector_type(8)))  float  v8f;

static __device__ __forceinline__ __bf16 cvt_bf16(float f) { return (__bf16)f; }

// ---- degree / normalization -------------------------------------------------

__global__ void k_fill(float* __restrict__ p, float v, int n) {
  int i = blockIdx.x * blockDim.x + threadIdx.x;
  if (i < n) p[i] = v;
}

__global__ void k_deg_edges(const int* __restrict__ src, const int* __restrict__ dst,
                            float* __restrict__ deg, int ne) {
  int e = blockIdx.x * blockDim.x + threadIdx.x;
  if (e < ne) {
    atomicAdd(&deg[src[e]], 1.0f);
    atomicAdd(&deg[dst[e]], 1.0f);
  }
}

__global__ void k_rsqrt(float* __restrict__ p, int n) {
  int i = blockIdx.x * blockDim.x + threadIdx.x;
  if (i < n) p[i] = rsqrtf(p[i]);
}

// ---- SpMM: agg = D^-1/2 (A + A^T + I) D^-1/2 @ feat  (feature dim fixed 128)

__global__ void k_spmm_init(const float* __restrict__ feat, const float* __restrict__ dinv,
                            float* __restrict__ agg, int nf) {
  int i = blockIdx.x * blockDim.x + threadIdx.x;
  if (i < nf) {
    float di = dinv[i >> 7];          // self-loop term: w = dinv[i]^2
    agg[i] = di * di * feat[i];
  }
}

// one thread per (edge, 4-feature group): float4 gathers, 8 coalesced f32 atomics
__global__ void k_spmm_edges(const int* __restrict__ src, const int* __restrict__ dst,
                             const float* __restrict__ feat, const float* __restrict__ dinv,
                             float* __restrict__ agg, int ne) {
  int t = blockIdx.x * blockDim.x + threadIdx.x;
  if (t >= (ne << 5)) return;
  int e = t >> 5, f4 = (t & 31) << 2;
  int s = src[e], d = dst[e];
  float w = dinv[s] * dinv[d];
  const float4 xs = *(const float4*)(feat + (s << 7) + f4);
  const float4 xd = *(const float4*)(feat + (d << 7) + f4);
  float* as = agg + (s << 7) + f4;
  float* ad = agg + (d << 7) + f4;
  atomicAdd(as + 0, w * xd.x); atomicAdd(as + 1, w * xd.y);
  atomicAdd(as + 2, w * xd.z); atomicAdd(as + 3, w * xd.w);
  atomicAdd(ad + 0, w * xs.x); atomicAdd(ad + 1, w * xs.y);
  atomicAdd(ad + 2, w * xs.z); atomicAdd(ad + 3, w * xs.w);
}

// ---- f32 -> bf16 conversion (row-major, 8 elems/thread) ---------------------

__global__ void k_cvt_bf16(const float* __restrict__ in, __bf16* __restrict__ out, int n8) {
  int t = blockIdx.x * blockDim.x + threadIdx.x;
  if (t >= n8) return;
  const float4 a = ((const float4*)in)[(size_t)t * 2];
  const float4 b = ((const float4*)in)[(size_t)t * 2 + 1];
  v8bf o;
  o[0] = cvt_bf16(a.x); o[1] = cvt_bf16(a.y); o[2] = cvt_bf16(a.z); o[3] = cvt_bf16(a.w);
  o[4] = cvt_bf16(b.x); o[5] = cvt_bf16(b.y); o[6] = cvt_bf16(b.z); o[7] = cvt_bf16(b.w);
  ((v8bf*)out)[t] = o;
}

// ---- pack B[K x Ncols] (f32 row-major) into per-lane WMMA fragment layout ---
// Bp layout: [ntile][kstep][lane][16 bf16]; lane gives col = 16*nt + (l&15),
// K range = 32*s + 16*(l>>4) + [0,16)  (B fragment: lanes 0-15 K 0..15, 16-31 K 16..31)

__global__ void k_pack_B(const float* __restrict__ B, __bf16* __restrict__ Bp,
                         int K, int Ncols) {
  int t = blockIdx.x * blockDim.x + threadIdx.x;
  int ksteps = K >> 5;
  int total = (Ncols >> 4) * ksteps * 32;
  if (t >= total) return;
  int lane = t & 31;
  int rest = t >> 5;
  int s  = rest % ksteps;
  int nt = rest / ksteps;
  int col   = (nt << 4) + (lane & 15);
  int kbase = (s << 5) + ((lane >> 4) << 4);
  v16bf o;
#pragma unroll
  for (int j = 0; j < 16; ++j) o[j] = cvt_bf16(B[(size_t)(kbase + j) * Ncols + col]);
  ((v16bf*)Bp)[t] = o;   // t == ((nt*ksteps + s)*32 + lane)
}

// ---- dense GEMM via v_wmma_f32_16x16x32_bf16 --------------------------------
// C[M x Ncols] = A[M x K](bf16 row-major) @ Bp(packed fragments); K % 32 == 0.
// One wave per 16x16 tile: blockIdx.x = M-tile, threadIdx.y = N-tile.
// Inner loop: two b128 A loads + one 32B B load + one WMMA.

template <bool RELU>
__global__ void k_gemm_wmma(const __bf16* __restrict__ A, const __bf16* __restrict__ Bp,
                            float* __restrict__ C, int K, int Ncols) {
  const int lane = threadIdx.x;        // full wave active -> EXEC all 1s
  const int half = lane >> 4;
  const int l15  = lane & 15;
  const int tm   = blockIdx.x * 16;
  const int ksteps = K >> 5;

  // A fragment: row = tm+l15; per step K in {16h..16h+16}
  const __bf16* arow = A + (size_t)(tm + l15) * K + (half << 4);
  const v16bf* bptr  = (const v16bf*)Bp + (size_t)threadIdx.y * ksteps * 32 + lane;

  v8f acc = {};
  for (int s = 0; s < ksteps; ++s) {
    v8bf alo = *(const v8bf*)(arow);
    v8bf ahi = *(const v8bf*)(arow + 8);
    v16bf a = __builtin_shufflevector(alo, ahi,
        0, 1, 2, 3, 4, 5, 6, 7, 8, 9, 10, 11, 12, 13, 14, 15);
    v16bf b = bptr[(size_t)s * 32];
    // 8 args: (neg_a, A, neg_b, B, c_mod, C, reuse_a, reuse_b)
    acc = __builtin_amdgcn_wmma_f32_16x16x32_bf16(
        false, a, false, b, (short)0, acc, false, false);
    arow += 32;
  }

  // D layout: VGPR i -> row (8*half + i), col = 16*threadIdx.y + l15
  float* cp = C + (size_t)(tm + (half << 3)) * Ncols + (threadIdx.y << 4) + l15;
#pragma unroll
  for (int i = 0; i < 8; ++i) {
    float v = acc[i];
    if (RELU) v = fmaxf(v, 0.0f);
    cp[(size_t)i * Ncols] = v;
  }
}

// ---- launch -----------------------------------------------------------------

extern "C" void kernel_launch(void* const* d_in, const int* in_sizes, int n_in,
                              void* d_out, int out_size, void* d_ws, size_t ws_size,
                              hipStream_t stream) {
  const float* x  = (const float*)d_in[0];
  const float* W0 = (const float*)d_in[1];
  const float* W1 = (const float*)d_in[2];
  const int*   ei = (const int*)d_in[3];
  const int* src = ei;            // edge_index[0]
  const int* dst = ei + NE;       // edge_index[1]
  float* out = (float*)d_out;

  char* ws = (char*)d_ws;
  size_t off = 0;
  float*  dinv = (float*)(ws + off);  off += ((size_t)NN * 4 + 255) & ~(size_t)255;
  float*  agg  = (float*)(ws + off);  off += ((size_t)NN * DF * 4 + 255) & ~(size_t)255;
  float*  h    = (float*)(ws + off);  off += ((size_t)NN * DF * 4 + 255) & ~(size_t)255;
  __bf16* abf  = (__bf16*)(ws + off); off += ((size_t)NN * DF * 2 + 255) & ~(size_t)255;
  __bf16* W0p  = (__bf16*)(ws + off); off += ((size_t)DF * DF * 2 + 255) & ~(size_t)255;
  __bf16* W1p  = (__bf16*)(ws + off);

  const int NF = NN * DF;          // 6.4M
  const int ET = NE << 5;          // 25.6M scatter threads
  const int N8 = NF >> 3;

  // pack weights into WMMA fragment layout (tiny, once per call)
  k_pack_B<<<(8 * 4 * 32 + 255) / 256, 256, 0, stream>>>(W0, W0p, DF, DF);
  k_pack_B<<<(4 * 4 * 32 + 255) / 256, 256, 0, stream>>>(W1, W1p, DF, NC);

  // normalization: deg = 1 + scatter, dinv = rsqrt(deg)
  k_fill      <<<(NN + 255) / 256, 256, 0, stream>>>(dinv, 1.0f, NN);
  k_deg_edges <<<(NE + 255) / 256, 256, 0, stream>>>(src, dst, dinv, NE);
  k_rsqrt     <<<(NN + 255) / 256, 256, 0, stream>>>(dinv, NN);

  // layer 0: agg = Ahat @ x ; h = relu(agg @ W0)
  k_spmm_init <<<(NF + 255) / 256, 256, 0, stream>>>(x, dinv, agg, NF);
  k_spmm_edges<<<(ET + 255) / 256, 256, 0, stream>>>(src, dst, x, dinv, agg, NE);
  k_cvt_bf16  <<<(N8 + 255) / 256, 256, 0, stream>>>(agg, abf, N8);
  k_gemm_wmma<true><<<dim3(NN / 16), dim3(32, DF / 16), 0, stream>>>(abf, W0p, h, DF, DF);

  // layer 1: agg = Ahat @ h ; out = agg @ W1
  k_spmm_init <<<(NF + 255) / 256, 256, 0, stream>>>(h, dinv, agg, NF);
  k_spmm_edges<<<(ET + 255) / 256, 256, 0, stream>>>(src, dst, h, dinv, agg, NE);
  k_cvt_bf16  <<<(N8 + 255) / 256, 256, 0, stream>>>(agg, abf, N8);
  k_gemm_wmma<false><<<dim3(NN / 16), dim3(32, NC / 16), 0, stream>>>(abf, W1p, out, DF, NC);
}
